// Model_43224550868041
// MI455X (gfx1250) — matrix operand order repaired and checked by reading, not compiled, not run on hardware
//
#include <hip/hip_runtime.h>
#include <hip/hip_bf16.h>

typedef __attribute__((ext_vector_type(16))) _Float16 v16h;
typedef __attribute__((ext_vector_type(8)))  _Float16 v8h;
typedef __attribute__((ext_vector_type(8)))  float    v8f;

#define N_NODES 50000
#define N_EDGES 800000
#define IN_CH   1490
#define KPAD1   1504   // 47 * 32
#define HID     800    // 25 * 32
#define OUT_CH  400

// ---------------------------------------------------------------------------
// f32 -> f16 conversion with K padding (pads k in [kin, kout) with zeros)
// ---------------------------------------------------------------------------
__global__ void cvt_f32_to_f16_pad(const float* __restrict__ in,
                                   _Float16* __restrict__ out,
                                   long long rows, int kin, int kout) {
    long long total = rows * (long long)kout;
    for (long long i = blockIdx.x * (long long)blockDim.x + threadIdx.x;
         i < total; i += (long long)gridDim.x * blockDim.x) {
        long long r = i / kout;
        int k = (int)(i - r * kout);
        out[i] = (k < kin) ? (_Float16)in[r * (long long)kin + k] : (_Float16)0.0f;
    }
}

// ---------------------------------------------------------------------------
// WMMA GEMM: C[M,N] = A[M,K] (f16, row-major) * B[N,K]^T (f16, row-major = W)
// One wave computes a 16x80 strip: 1 A fragment x 5 B fragments -> 5 accum.
// M % 16 == 0, N % 80 == 0, K % 32 == 0 (guaranteed by padding).
// ---------------------------------------------------------------------------
__global__ void __launch_bounds__(128)
wmma_gemm_f16(const _Float16* __restrict__ A, const _Float16* __restrict__ B,
              float* __restrict__ C, int M, int N, int K,
              int mtiles, int totalWaves) {
    const int lane  = threadIdx.x & 31;
    const int waveG = blockIdx.x * (blockDim.x >> 5) + (threadIdx.x >> 5);
    if (waveG >= totalWaves) return;             // uniform per wave

    const int mtile = waveG % mtiles;
    const int group = waveG / mtiles;
    const int m0 = mtile * 16;
    const int n0 = group * 80;

    const int col = lane & 15;                   // M-row for A, N-col for B/C
    const int hi  = lane >> 4;                   // half-wave select

    const size_t arow = (size_t)(m0 + col) * (size_t)K;

    v8f acc[5] = {};

    for (int k = 0; k < K; k += 32) {
        // A fragment (16x32 f16): lanes0-15 K {k..k+15},
        // lanes16-31 K {k+16..k+31} (contiguous, matches B).
        union { v16h v; v8h h[2]; } a;
        a.h[0] = *(const v8h*)(A + arow + k + hi * 16);
        a.h[1] = *(const v8h*)(A + arow + k + hi * 16 + 8);

#pragma unroll
        for (int j = 0; j < 5; ++j) {
            // B fragment (32x16): per lane column n0+j*16+col, 16 contiguous
            // K-halves selected by half-wave -> single 32B load.
            const v16h b = *(const v16h*)(B + (size_t)(n0 + j * 16 + col) * (size_t)K
                                            + k + hi * 16);
            acc[j] = __builtin_amdgcn_wmma_f32_16x16x32_f16(
                false, a.v, false, b, (short)0, acc[j], false, false);
        }
    }

    // D layout: VGPR r -> row m0 + hi*8 + r, col n0 + j*16 + col
#pragma unroll
    for (int j = 0; j < 5; ++j) {
        float* cp = C + (size_t)(m0 + hi * 8) * (size_t)N + n0 + j * 16 + col;
#pragma unroll
        for (int r = 0; r < 8; ++r)
            cp[(size_t)r * (size_t)N] = acc[j][r];
    }
}

// ---------------------------------------------------------------------------
// Degree / normalization
// ---------------------------------------------------------------------------
__global__ void deg_init(float* __restrict__ deg, int n) {
    int i = blockIdx.x * blockDim.x + threadIdx.x;
    if (i < n) deg[i] = 1.0f;                    // self-loop contribution
}

__global__ void deg_count(const int* __restrict__ dst, float* __restrict__ deg, int e) {
    int i = blockIdx.x * blockDim.x + threadIdx.x;
    if (i < e) atomicAdd(&deg[dst[i]], 1.0f);
}

__global__ void deg_rsqrt(float* __restrict__ deg, int n) {
    int i = blockIdx.x * blockDim.x + threadIdx.x;
    if (i < n) deg[i] = rsqrtf(deg[i]);          // deg >= 1 always
}

// ---------------------------------------------------------------------------
// Aggregation: out[n][:] starts with self-loop term dinv[n]^2 * h[n][:]
// ---------------------------------------------------------------------------
__global__ void self_loop_init(const float* __restrict__ h,
                               const float* __restrict__ dinv,
                               float* __restrict__ out, long long total, int C) {
    for (long long i = blockIdx.x * (long long)blockDim.x + threadIdx.x;
         i < total; i += (long long)gridDim.x * blockDim.x) {
        long long node = i / C;
        float w = dinv[node];
        out[i] = w * w * h[i];
    }
}

// one block per edge: coalesced gather of h[src][:], atomic scatter to out[dst][:]
__global__ void agg_edges(const int* __restrict__ src, const int* __restrict__ dst,
                          const float* __restrict__ dinv,
                          const float* __restrict__ h, float* __restrict__ out, int C) {
    int e = blockIdx.x;
    int s = src[e];
    int d = dst[e];
    float w = dinv[s] * dinv[d];
    const float* hp = h + (size_t)s * C;
    float* op = out + (size_t)d * C;
    for (int c = threadIdx.x; c < C; c += blockDim.x)
        atomicAdd(&op[c], w * hp[c]);
}

__global__ void bias_relu(float* __restrict__ x, const float* __restrict__ b,
                          long long total, int C) {
    for (long long i = blockIdx.x * (long long)blockDim.x + threadIdx.x;
         i < total; i += (long long)gridDim.x * blockDim.x) {
        float v = x[i] + b[i % C];
        x[i] = v > 0.0f ? v : 0.0f;
    }
}

// ---------------------------------------------------------------------------
// Host-side orchestration
// ---------------------------------------------------------------------------
extern "C" void kernel_launch(void* const* d_in, const int* in_sizes, int n_in,
                              void* d_out, int out_size, void* d_ws, size_t ws_size,
                              hipStream_t stream) {
    const float* x  = (const float*)d_in[0];
    const int*   ei = (const int*)d_in[1];          // [2, E]; row0=src, row1=dst
    const float* W1 = (const float*)d_in[2];        // [HID, IN_CH]
    const float* b1 = (const float*)d_in[3];
    const float* W2 = (const float*)d_in[4];        // [OUT, HID]
    const float* b2 = (const float*)d_in[5];
    float* out = (float*)d_out;                     // [N_NODES, OUT_CH]

    const int* src = ei;
    const int* dst = ei + N_EDGES;

    // Workspace layout (bytes, all 256-aligned)
    char* ws = (char*)d_ws;
    const size_t OFF_XH   = 0;                                  // 50000*1504*2 = 150,400,000
    const size_t OFF_W1H  = OFF_XH  + (size_t)N_NODES * KPAD1 * 2;
    const size_t OFF_W2H  = OFF_W1H + (size_t)HID * KPAD1 * 2;  // 2,406,400
    const size_t OFF_H1   = OFF_W2H + (size_t)OUT_CH * HID * 2; // 640,000
    const size_t OFF_AGG1 = OFF_H1  + (size_t)N_NODES * HID * 4;
    const size_t OFF_DINV = OFF_AGG1 + (size_t)N_NODES * HID * 4;

    _Float16* xh   = (_Float16*)(ws + OFF_XH);
    _Float16* w1h  = (_Float16*)(ws + OFF_W1H);
    _Float16* w2h  = (_Float16*)(ws + OFF_W2H);
    float*    h1   = (float*)(ws + OFF_H1);
    float*    agg1 = (float*)(ws + OFF_AGG1);
    float*    dinv = (float*)(ws + OFF_DINV);
    _Float16* h2h  = (_Float16*)(ws + OFF_XH);      // reuse xh region (dead after GEMM1)
    float*    h3   = (float*)(ws + OFF_H1);         // reuse h1 region (dead after agg1)

    const int T = 256;
    const int GS = 4096;                            // grid-stride blocks

    // --- normalization ---
    deg_init<<<(N_NODES + T - 1) / T, T, 0, stream>>>(dinv, N_NODES);
    deg_count<<<(N_EDGES + T - 1) / T, T, 0, stream>>>(dst, dinv, N_EDGES);
    deg_rsqrt<<<(N_NODES + T - 1) / T, T, 0, stream>>>(dinv, N_NODES);

    // --- convert inputs to f16 (K padded to multiple of 32) ---
    cvt_f32_to_f16_pad<<<GS, T, 0, stream>>>(x,  xh,  N_NODES, IN_CH, KPAD1);
    cvt_f32_to_f16_pad<<<GS, T, 0, stream>>>(W1, w1h, HID,     IN_CH, KPAD1);
    cvt_f32_to_f16_pad<<<GS, T, 0, stream>>>(W2, w2h, OUT_CH,  HID,   HID);

    // --- layer 1: GEMM (WMMA f16 -> f32) ---
    {
        int mtiles = N_NODES / 16;                  // 3125
        int ngroups = HID / 80;                     // 10
        int waves = mtiles * ngroups;
        int blocks = (waves + 3) / 4;               // 4 waves / 128-thread block
        wmma_gemm_f16<<<blocks, 128, 0, stream>>>(xh, w1h, h1,
                                                  N_NODES, HID, KPAD1, mtiles, waves);
    }
    // --- layer 1: aggregation + bias + relu ---
    self_loop_init<<<GS, T, 0, stream>>>(h1, dinv, agg1, (long long)N_NODES * HID, HID);
    agg_edges<<<N_EDGES, T, 0, stream>>>(src, dst, dinv, h1, agg1, HID);
    bias_relu<<<GS, T, 0, stream>>>(agg1, b1, (long long)N_NODES * HID, HID);

    // --- layer 2 ---
    cvt_f32_to_f16_pad<<<GS, T, 0, stream>>>(agg1, h2h, N_NODES, HID, HID);
    {
        int mtiles = N_NODES / 16;                  // 3125
        int ngroups = OUT_CH / 80;                  // 5
        int waves = mtiles * ngroups;
        int blocks = (waves + 3) / 4;
        wmma_gemm_f16<<<blocks, 128, 0, stream>>>(h2h, w2h, h3,
                                                  N_NODES, OUT_CH, HID, mtiles, waves);
    }
    self_loop_init<<<GS, T, 0, stream>>>(h3, dinv, out, (long long)N_NODES * OUT_CH, OUT_CH);
    agg_edges<<<N_EDGES, T, 0, stream>>>(src, dst, dinv, h3, out, OUT_CH);
    bias_relu<<<GS, T, 0, stream>>>(out, b2, (long long)N_NODES * OUT_CH, OUT_CH);
}
